// NetworkLayer_30700426232150
// MI455X (gfx1250) — compile-verified, matrix operand order checked by reading, not run on hardware
//
#include <hip/hip_runtime.h>
#include <hip/hip_bf16.h>

// ---------------------------------------------------------------------------
// MI455X (gfx1250): all GEMM stages + both attention matmuls via
// v_wmma_f32_16x16x32_f16 (f16 in, f32 accumulate). Inputs/weights are
// pre-converted to f16 once; GEMM tiles are filled with double-buffered
// GLOBAL_LOAD_ASYNC_TO_LDS_B128 (ASYNCcnt-tracked global->LDS copies), so
// the HBM fill of tile k+1 overlaps the WMMA work on tile k.
// ---------------------------------------------------------------------------

typedef __attribute__((ext_vector_type(16))) _Float16 v16h;
typedef __attribute__((ext_vector_type(8)))  _Float16 v8h;
typedef __attribute__((ext_vector_type(8)))  float    v8f;

#define NTOK   262144LL
#define DMODEL 512
#define DPOS   256
#define NHEAD  8
#define DFF    2048
#define NGRP   2048
#define NCHUNK 8
#define CHTOK  (NTOK / NCHUNK)   // 32768 tokens per MLP chunk

// --------------------------- WMMA fragment helpers -------------------------
// 16-bit A 16x32 layout (ISA 7.12.2): lane l holds row M=l%16; halves
// [k0,k0+8) and [k0+16,k0+24), k0 = (l<16 ? 0 : 8). stride in halves,
// stride%8==0 -> both reads are 16B aligned ds_load_b128.
__device__ __forceinline__ v16h ldfrag(const _Float16* base, int stride, int lane) {
    const _Float16* p = base + (size_t)(lane & 15) * stride + ((lane >> 4) << 3);
    v8h lo = *(const v8h*)(p);
    v8h hi = *(const v8h*)(p + 16);
    return __builtin_shufflevector(lo, hi, 0,1,2,3,4,5,6,7,8,9,10,11,12,13,14,15);
}

__device__ __forceinline__ v8f wmma_f16(v16h a, v16h b, v8f c) {
    return __builtin_amdgcn_wmma_f32_16x16x32_f16(false, a, false, b, (short)0, c,
                                                  false, false);
}
// C/D 16x16 f32 layout: element e of lane l -> row = e + 8*(l>>4), col = l&15.

// ----------------------- async global->LDS tile fill -----------------------
// GLOBAL_LOAD_ASYNC_TO_LDS_B128, GVS mode: mem = SGPR64 + VGPR32 + ioffset;
// LDS dest = wave LDS base + VGPR (low 32 bits of the generic shared ptr).
__device__ __forceinline__ void async_cp16(void* lds_ptr, unsigned gl_byte_off,
                                           const void* base) {
    unsigned lds_byte = (unsigned)(unsigned long long)lds_ptr;
    asm volatile("global_load_async_to_lds_b128 %0, %1, %2"
                 :: "v"(lds_byte), "v"(gl_byte_off), "s"(base) : "memory");
}
__device__ __forceinline__ void async_wait0() {
    asm volatile("s_wait_asynccnt 0x0" ::: "memory");
}

// ------------------------------- GEMM kernel -------------------------------
// C[M,N] = act( A[M,K] @ B[N,K]^T + bias[N] (+ resid[M,N]) ), all-f16 inputs.
// Block tile 128x128, 256 threads (8 waves); wave tile 32x64 (2x4 frags):
// 8 WMMAs per 6 fragment loads per k-step. Double-buffered async LDS fills.
// Optional K-split A (concat[x,pe] for MLP-A).
#define BM  128
#define BN  128
#define GK  32
#define GLD 40   // padded LDS row stride in halves (80B: 16B-aligned)

template<bool SPLIT, bool RELU, bool OUT_HALF, bool RESID>
__global__ __launch_bounds__(256)
void gemm_wmma(const _Float16* __restrict__ A1, const _Float16* __restrict__ A2, int K1,
               const _Float16* __restrict__ Bw, const float* __restrict__ bias,
               const float* __restrict__ resid, void* __restrict__ Cp,
               int M, int N, int K)
{
    __shared__ _Float16 As[2][BM * GLD];
    __shared__ _Float16 Bs[2][BN * GLD];

    const int tid  = threadIdx.x;
    const int lane = tid & 31;
    const int w    = tid >> 5;
    const int n0   = blockIdx.x * BN;
    const int m0   = blockIdx.y * BM;
    const int wm   = (w & 3) * 32;          // 4 waves down M
    const int wn   = (w >> 2) * 64;         // 2 waves across N

    const int row = tid >> 1;               // 0..127: tile row this thread fills
    const int seg = (tid & 1) * 16;         // halves 0..15 / 16..31

    // async fill of one 128x32-half A tile + one B tile into buffer pb
    auto fill = [&](int kk, int pb) {
        const _Float16* base; int ld;
        if (!SPLIT || kk < K1) { base = A1 + (size_t)m0 * (SPLIT ? K1 : K) + kk;
                                 ld = SPLIT ? K1 : K; }
        else                   { base = A2 + (size_t)m0 * (K - K1) + (kk - K1);
                                 ld = K - K1; }
        const unsigned ga = (unsigned)(row * ld + seg) * 2u;
        async_cp16(&As[pb][row * GLD + seg],     ga,       base);
        async_cp16(&As[pb][row * GLD + seg + 8], ga + 16u, base);

        const _Float16* bbase = Bw + (size_t)n0 * K + kk;
        const unsigned gb = (unsigned)(row * K + seg) * 2u;
        async_cp16(&Bs[pb][row * GLD + seg],     gb,       bbase);
        async_cp16(&Bs[pb][row * GLD + seg + 8], gb + 16u, bbase);
    };

    v8f acc[2][4] = {};

    int pb = 0;
    fill(0, 0);
    async_wait0();
    __syncthreads();

    for (int kk = 0; kk < K; kk += GK) {
        const int nxt = kk + GK;
        if (nxt < K) fill(nxt, pb ^ 1);     // prefetch next tile, no wait

        const v16h a0 = ldfrag(&As[pb][(wm)      * GLD], GLD, lane);
        const v16h a1 = ldfrag(&As[pb][(wm + 16) * GLD], GLD, lane);
        #pragma unroll
        for (int j = 0; j < 4; ++j) {
            const v16h bj = ldfrag(&Bs[pb][(wn + j * 16) * GLD], GLD, lane);
            acc[0][j] = wmma_f16(a0, bj, acc[0][j]);
            acc[1][j] = wmma_f16(a1, bj, acc[1][j]);
        }

        async_wait0();                      // our prefetch writes have landed
        __syncthreads();                    // everyone done reading buf pb
        pb ^= 1;
    }

    // ---- epilogue ----
    const int ce = lane & 15;
    const int rh = (lane >> 4) << 3;
    #pragma unroll
    for (int i = 0; i < 2; ++i) {
        #pragma unroll
        for (int j = 0; j < 4; ++j) {
            const int colg = n0 + wn + j * 16 + ce;
            const float bv = bias[colg];
            #pragma unroll
            for (int e = 0; e < 8; ++e) {
                const int rowg = m0 + wm + i * 16 + e + rh;
                float v = acc[i][j][e] + bv;
                if constexpr (RESID) v += resid[(size_t)rowg * (size_t)N + colg];
                if constexpr (RELU)  v = v > 0.f ? v : 0.f;
                if constexpr (OUT_HALF)
                    ((_Float16*)Cp)[(size_t)rowg * (size_t)N + colg] = (_Float16)v;
                else
                    ((float*)Cp)[(size_t)rowg * (size_t)N + colg] = v;
            }
        }
    }
}

// --------------------------- f32 -> f16 conversion -------------------------
__global__ __launch_bounds__(256)
void cvt_f32_f16(const float* __restrict__ s, _Float16* __restrict__ d, long long n)
{
    const long long i = ((long long)blockIdx.x * 256 + threadIdx.x) * 8;
    if (i >= n) return;
    const float4 a = *(const float4*)(s + i);
    const float4 b = *(const float4*)(s + i + 4);
    v8h o;
    o[0] = (_Float16)a.x; o[1] = (_Float16)a.y; o[2] = (_Float16)a.z; o[3] = (_Float16)a.w;
    o[4] = (_Float16)b.x; o[5] = (_Float16)b.y; o[6] = (_Float16)b.z; o[7] = (_Float16)b.w;
    *(v8h*)(d + i) = o;
}

// ---------------------------- Attention kernel -----------------------------
// One workgroup (256 threads, 8 waves) per (group a, head h).
//   S = Q @ K^T (128x128, K=32 -> 1 WMMA/tile); transform
//   1/(sqrt(max(q2+k2-2s,1e-12))+1), causal mask col>row -> 0; O = S @ V.
// Q/K tiles + norms are union-overlaid by the f16 score tile (~51KB LDS).
#define SST 136   // padded score/Vt row stride in halves (272B: 16B-aligned)

__global__ __launch_bounds__(256)
void attn_wmma(const _Float16* __restrict__ Qh, const _Float16* __restrict__ Kh,
               const _Float16* __restrict__ Vh, _Float16* __restrict__ Oh)
{
    __shared__ union {
        struct {
            _Float16 Qs[128 * GLD];
            _Float16 Ks[128 * GLD];
            float    q2[128];
            float    k2[128];
        } p;
        _Float16 Ss[128 * SST];
    } u;
    __shared__ _Float16 Vt[64 * SST];

    const int a = blockIdx.x, h = blockIdx.y;
    const int tid = threadIdx.x, lane = tid & 31, w = tid >> 5;
    const size_t tokBase = (size_t)a * 128;

    // ---- phase 1: load Q,K (128x32) and V^T (64x128), compute |q|^2,|k|^2 ----
    {
        const int c = tid >> 1, seg = (tid & 1) * 16;
        const _Float16* gq = Qh + (tokBase + c) * DPOS + h * 32 + seg;
        const _Float16* gk = Kh + (tokBase + c) * DPOS + h * 32 + seg;
        *(v8h*)&u.p.Qs[c * GLD + seg]     = *(const v8h*)(gq);
        *(v8h*)&u.p.Qs[c * GLD + seg + 8] = *(const v8h*)(gq + 8);
        *(v8h*)&u.p.Ks[c * GLD + seg]     = *(const v8h*)(gk);
        *(v8h*)&u.p.Ks[c * GLD + seg + 8] = *(const v8h*)(gk + 8);

        const int dseg = (tid & 1) * 32;
        const _Float16* gv = Vh + (tokBase + c) * DMODEL + h * 64 + dseg;
        v8h v0 = *(const v8h*)(gv);
        v8h v1 = *(const v8h*)(gv + 8);
        v8h v2 = *(const v8h*)(gv + 16);
        v8h v3 = *(const v8h*)(gv + 24);
        #pragma unroll
        for (int d = 0; d < 8; ++d) {
            Vt[(dseg + d)      * SST + c] = v0[d];
            Vt[(dseg + 8 + d)  * SST + c] = v1[d];
            Vt[(dseg + 16 + d) * SST + c] = v2[d];
            Vt[(dseg + 24 + d) * SST + c] = v3[d];
        }
    }
    __syncthreads();
    if (tid < 128) {
        float s = 0.f;
        #pragma unroll
        for (int d = 0; d < 32; ++d) { const float q = (float)u.p.Qs[tid * GLD + d]; s += q * q; }
        u.p.q2[tid] = s;
    } else {
        const int c = tid - 128;
        float s = 0.f;
        #pragma unroll
        for (int d = 0; d < 32; ++d) { const float k = (float)u.p.Ks[c * GLD + d]; s += k * k; }
        u.p.k2[c] = s;
    }
    __syncthreads();

    // ---- phase 2: scores (each wave owns 16 rows x 128 cols) ----
    const int rbase = w * 16;
    const int ce = lane & 15;
    const int rh = (lane >> 4) << 3;

    const v16h aq = ldfrag(&u.p.Qs[rbase * GLD], GLD, lane);
    v8f sc[8];
    #pragma unroll
    for (int nt = 0; nt < 8; ++nt) {
        const v16h bk = ldfrag(&u.p.Ks[nt * 16 * GLD], GLD, lane);
        v8f z = {};
        sc[nt] = wmma_f16(aq, bk, z);
    }
    float q2r[8];
    #pragma unroll
    for (int e = 0; e < 8; ++e) q2r[e] = u.p.q2[rbase + e + rh];
    #pragma unroll
    for (int nt = 0; nt < 8; ++nt) {
        const int col = nt * 16 + ce;
        const float k2c = u.p.k2[col];
        #pragma unroll
        for (int e = 0; e < 8; ++e) {
            const int row = rbase + e + rh;
            float d2 = q2r[e] + k2c - 2.f * sc[nt][e];
            d2 = fmaxf(d2, 1e-12f);
            const float dist = sqrtf(d2) + 1.f;
            sc[nt][e] = (col > row) ? 0.f : (1.f / dist);
        }
    }
    __syncthreads();                       // Qs/Ks/q2/k2 dead from here on
    #pragma unroll
    for (int nt = 0; nt < 8; ++nt)
        #pragma unroll
        for (int e = 0; e < 8; ++e)
            u.Ss[(rbase + e + rh) * SST + nt * 16 + ce] = (_Float16)sc[nt][e];
    __syncthreads();

    // ---- phase 3: O = S @ V (wave: 16 rows x 64 cols, K=128 in 4 steps) ----
    v8f oa[4] = {};
    #pragma unroll
    for (int ks = 0; ks < 4; ++ks) {
        const int kk = ks * 32;
        const v16h as = ldfrag(&u.Ss[rbase * SST + kk], SST, lane);
        #pragma unroll
        for (int j = 0; j < 4; ++j) {
            const v16h bv = ldfrag(&Vt[(j * 16) * SST + kk], SST, lane);
            oa[j] = wmma_f16(as, bv, oa[j]);
        }
    }
    #pragma unroll
    for (int j = 0; j < 4; ++j)
        #pragma unroll
        for (int e = 0; e < 8; ++e) {
            const int row = rbase + e + rh;
            const int col = j * 16 + ce;
            Oh[(tokBase + row) * DMODEL + h * 64 + col] = (_Float16)oa[j][e];
        }
}

// ------------------------------ host launcher ------------------------------
static inline void cvt(const float* s, _Float16* d, long long n, hipStream_t st) {
    cvt_f32_f16<<<dim3((unsigned)(n / 8 / 256)), dim3(256), 0, st>>>(s, d, n);
}

extern "C" void kernel_launch(void* const* d_in, const int* in_sizes, int n_in,
                              void* d_out, int out_size, void* d_ws, size_t ws_size,
                              hipStream_t stream) {
    const float* x   = (const float*)d_in[0];   // [NTOK, 512]
    const float* pe  = (const float*)d_in[1];   // [NTOK, 256]
    const float* Wq  = (const float*)d_in[2];   const float* bq  = (const float*)d_in[3];
    const float* Wk  = (const float*)d_in[4];   const float* bk  = (const float*)d_in[5];
    const float* Wv  = (const float*)d_in[6];   const float* bv  = (const float*)d_in[7];
    const float* Wf1 = (const float*)d_in[8];   const float* bf1 = (const float*)d_in[9];
    const float* Wf2 = (const float*)d_in[10];  const float* bf2 = (const float*)d_in[11];
    const float* Wa1 = (const float*)d_in[12];  const float* ba1 = (const float*)d_in[13];
    const float* Wa2 = (const float*)d_in[14];  const float* ba2 = (const float*)d_in[15];
    float* out = (float*)d_out;

    // f16 workspace layout (~1.5 GB total)
    _Float16* p = (_Float16*)d_ws;
    _Float16* xh   = p;  p += NTOK * DMODEL;               // [NTOK,512]
    _Float16* peh  = p;  p += NTOK * DPOS;                 // [NTOK,256]
    _Float16* Wa1h = p;  p += (size_t)DFF * (DMODEL + DPOS);
    _Float16* Wa2h = p;  p += (size_t)DPOS * DFF;
    _Float16* Wqh  = p;  p += (size_t)DPOS * DPOS;
    _Float16* Wkh  = p;  p += (size_t)DPOS * DPOS;
    _Float16* Wvh  = p;  p += (size_t)DMODEL * DMODEL;
    _Float16* Wf1h = p;  p += (size_t)DFF * DMODEL;
    _Float16* Wf2h = p;  p += (size_t)DMODEL * DFF;
    _Float16* hid  = p;  p += (size_t)CHTOK * DFF;         // reused per chunk
    _Float16* pep  = p;  p += NTOK * DPOS;                 // pe + ff
    _Float16* Qh   = p;  p += NTOK * DPOS;
    _Float16* Khb  = p;  p += NTOK * DPOS;
    _Float16* Vhb  = p;  p += NTOK * DMODEL;
    _Float16* Oh   = p;  p += NTOK * DMODEL;

    // ---- one-time f32 -> f16 conversions (inputs + weights) ----
    cvt(x,   xh,   NTOK * DMODEL, stream);
    cvt(pe,  peh,  NTOK * DPOS,   stream);
    cvt(Wa1, Wa1h, (long long)DFF * (DMODEL + DPOS), stream);
    cvt(Wa2, Wa2h, (long long)DPOS * DFF,    stream);
    cvt(Wq,  Wqh,  (long long)DPOS * DPOS,   stream);
    cvt(Wk,  Wkh,  (long long)DPOS * DPOS,   stream);
    cvt(Wv,  Wvh,  (long long)DMODEL * DMODEL, stream);
    cvt(Wf1, Wf1h, (long long)DFF * DMODEL,  stream);
    cvt(Wf2, Wf2h, (long long)DMODEL * DFF,  stream);

    const dim3 blk(256);

    // --- MLP-A (chunked): hid = relu([x|pe]@Wa1^T+ba1); pep = pe + hid@Wa2^T+ba2
    for (int c = 0; c < NCHUNK; ++c) {
        const long long t0 = (long long)c * CHTOK;
        gemm_wmma<true, true, true, false>
            <<<dim3(DFF / BN, CHTOK / BM), blk, 0, stream>>>(
                xh + t0 * DMODEL, peh + t0 * DPOS, DMODEL,
                Wa1h, ba1, nullptr, hid, CHTOK, DFF, DMODEL + DPOS);
        gemm_wmma<false, false, true, true>
            <<<dim3(DPOS / BN, CHTOK / BM), blk, 0, stream>>>(
                hid, nullptr, 0, Wa2h, ba2, pe + t0 * DPOS,
                pep + t0 * DPOS, CHTOK, DPOS, DFF);
    }

    // --- Q/K from pep, V from xh ---
    gemm_wmma<false, false, true, false>
        <<<dim3(DPOS / BN, NTOK / BM), blk, 0, stream>>>(
            pep, nullptr, 0, Wqh, bq, nullptr, Qh, (int)NTOK, DPOS, DPOS);
    gemm_wmma<false, false, true, false>
        <<<dim3(DPOS / BN, NTOK / BM), blk, 0, stream>>>(
            pep, nullptr, 0, Wkh, bk, nullptr, Khb, (int)NTOK, DPOS, DPOS);
    gemm_wmma<false, false, true, false>
        <<<dim3(DMODEL / BN, NTOK / BM), blk, 0, stream>>>(
            xh, nullptr, 0, Wvh, bv, nullptr, Vhb, (int)NTOK, DMODEL, DMODEL);

    // --- euclidean-attention over 2048 groups x 8 heads ---
    attn_wmma<<<dim3(NGRP, NHEAD), dim3(256), 0, stream>>>(Qh, Khb, Vhb, Oh);

    // --- MLP-F (chunked): out = relu(O@Wf1^T+bf1) @ Wf2^T + bf2
    for (int c = 0; c < NCHUNK; ++c) {
        const long long t0 = (long long)c * CHTOK;
        gemm_wmma<false, true, true, false>
            <<<dim3(DFF / BN, CHTOK / BM), blk, 0, stream>>>(
                Oh + t0 * DMODEL, nullptr, 0, Wf1h, bf1, nullptr,
                hid, CHTOK, DFF, DMODEL);
        gemm_wmma<false, false, false, false>
            <<<dim3(DMODEL / BN, CHTOK / BM), blk, 0, stream>>>(
                hid, nullptr, 0, Wf2h, bf2, nullptr,
                out + t0 * DMODEL, CHTOK, DMODEL, DFF);
    }
}
